// GRU_76046690943149
// MI455X (gfx1250) — matrix operand order repaired and checked by reading, not compiled, not run on hardware
//
#include <hip/hip_runtime.h>
#include <stdint.h>
#include <math.h>

// Problem dims (compile-time constants for this workload)
constexpr int B_ = 64;
constexpr int S_ = 512;
constexpr int I_ = 512;
constexpr int H_ = 1024;
constexpr int G_ = 3 * H_;          // 3072 gate columns

// LDS tile geometry for the persistent scan kernel:
// 3 gates x 16 rows x 1024 halves, rows padded to 1032 halves
// (516 dwords == 4 mod 64 banks -> 16 columns land on distinct banks).
constexpr int LDS_ROW   = 1032;                   // halves per padded row
constexpr int LDS_ROWS  = 48;                     // 3 gates * 16 rows
constexpr int LDS_BYTES = LDS_ROWS * LDS_ROW * 2; // 99,072 B (< 320KB/WGP)

// ---------------------------------------------------------------------------
// WMMA vector types (gfx1250, wave32)
// ---------------------------------------------------------------------------
typedef __attribute__((ext_vector_type(16))) __bf16   v16bf;
typedef __attribute__((ext_vector_type(8)))  float    v8f;
typedef __attribute__((ext_vector_type(4)))  uint32_t v4u;

union TileU {
    v4u      q[2];
    uint32_t u[8];
    v16bf    v;
};

// fp32 -> bf16 round-to-nearest-even
__device__ __forceinline__ uint16_t f2bf(float f) {
    uint32_t u = __float_as_uint(f);
    uint32_t r = (u + 0x7FFFu + ((u >> 16) & 1u)) >> 16;
    return (uint16_t)r;
}

__device__ __forceinline__ float sigmoidf_(float x) {
    return 1.0f / (1.0f + __expf(-x));
}

// A-matrix 16x32 bf16 tile (M rows x K), row-major global source.
// Lanes 0-15: row M = lane, K = k0..k0+15 contiguous.
// Lanes 16-31: same rows, K += 16.
__device__ __forceinline__ v16bf load_tileA(const uint16_t* p, int row0, int ld,
                                            int k0, int lane) {
    int row = row0 + (lane & 15);
    int k   = k0 + ((lane >> 4) << 4);
    const uint16_t* q = p + (size_t)row * ld + k;
    TileU t;
    t.q[0] = *(const v4u*)(q);
    t.q[1] = *(const v4u*)(q + 8);
    return t.v;
}

// B-matrix 32x16 bf16 tile (K x N) from row-major weights (col n == row n).
// Lanes 0-15: col N = lane, K = k0..k0+15 contiguous; lanes 16-31: K += 16.
__device__ __forceinline__ v16bf load_tileB(const uint16_t* p, int col0, int ld,
                                            int k0, int lane) {
    int col = col0 + (lane & 15);
    int k   = k0 + ((lane >> 4) << 4);
    const uint16_t* q = p + (size_t)col * ld + k;
    TileU t;
    t.q[0] = *(const v4u*)(q);
    t.q[1] = *(const v4u*)(q + 8);
    return t.v;
}

// Same B-tile gather but from an LDS-resident padded tile. `q` already points
// at this lane's (col,k-chunk) base; per K step advance by k0 halves.
__device__ __forceinline__ v16bf lds_tileB(const uint16_t* q) {
    TileU t;
    t.q[0] = *(const v4u*)(q);
    t.q[1] = *(const v4u*)(q + 8);
    return t.v;
}

#define WMMA_BF16(a, b, c) \
    __builtin_amdgcn_wmma_f32_16x16x32_bf16(false, (a), false, (b), (short)0, (c), false, false)

// ---------------------------------------------------------------------------
// Small utility kernels
// ---------------------------------------------------------------------------
__global__ void k_f32_to_bf16(const float* __restrict__ in,
                              uint16_t* __restrict__ out, int n) {
    int i = blockIdx.x * blockDim.x + threadIdx.x;
    if (i < n) out[i] = f2bf(in[i]);
}

__global__ void k_init_h(const float* __restrict__ h0,
                         float* __restrict__ hf, uint16_t* __restrict__ hb, int n) {
    int i = blockIdx.x * blockDim.x + threadIdx.x;
    if (i < n) {
        float v = h0[i];
        hf[i] = v;
        hb[i] = f2bf(v);
    }
}

__global__ void k_copy_f32(const float* __restrict__ src,
                           float* __restrict__ dst, int n) {
    int i = blockIdx.x * blockDim.x + threadIdx.x;
    if (i < n) dst[i] = src[i];
}

__global__ void k_zero_u32(unsigned int* p, int n) {
    int i = blockIdx.x * blockDim.x + threadIdx.x;
    if (i < n) p[i] = 0u;
}

// ---------------------------------------------------------------------------
// Input projection: gates[s][b][g] = inputs[b,s,:] . w_ih[g,:] + bias_ih[g]
// One wave computes a 16(M) x 64(N) strip; A tile reused across 4 N-tiles.
// ---------------------------------------------------------------------------
__global__ void k_xgemm(const uint16_t* __restrict__ xbf,
                        const uint16_t* __restrict__ wih,
                        const float* __restrict__ bias_ih,
                        float* __restrict__ gates) {
    const int lane = threadIdx.x & 31;
    const int wid  = blockIdx.x * (blockDim.x >> 5) + (threadIdx.x >> 5);
    const int mtile = wid / (G_ / 64);          // 0 .. 2047
    const int ng    = wid % (G_ / 64);          // 0 .. 47
    const int m0 = mtile * 16;
    const int n0 = ng * 64;

    v8f acc[4] = {};
    for (int k0 = 0; k0 < I_; k0 += 32) {
        // speculative prefetch of the A stream two K-tiles ahead
        __builtin_prefetch(xbf + (size_t)(m0 + (lane & 15)) * I_ + k0 + 64, 0, 1);
        v16bf a = load_tileA(xbf, m0, I_, k0, lane);
#pragma unroll
        for (int t = 0; t < 4; ++t) {
            v16bf b = load_tileB(wih, n0 + t * 16, I_, k0, lane);
            acc[t] = WMMA_BF16(a, b, acc[t]);
        }
    }

    const int nlane = lane & 15;
    const int mh    = (lane >> 4) << 3;
#pragma unroll
    for (int t = 0; t < 4; ++t) {
        const int g = n0 + t * 16 + nlane;
        const float bias = bias_ih[g];
#pragma unroll
        for (int r = 0; r < 8; ++r) {
            int row = m0 + mh + r;              // row = b*S + s
            int bb  = row / S_;
            int ss  = row % S_;
            gates[((size_t)ss * B_ + bb) * G_ + g] = acc[t][r] + bias;
        }
    }
}

// ---------------------------------------------------------------------------
// Device-wide barrier: monotonic counter, one arrival per block per phase.
// Counter is zeroed by k_zero_u32 at the start of every kernel_launch, so the
// persistent kernel is replay-deterministic under graph capture.
// ---------------------------------------------------------------------------
__device__ __forceinline__ void grid_sync(unsigned int* bar, unsigned int target) {
    __syncthreads();
    if (threadIdx.x == 0) {
        __threadfence();                        // make h writes visible device-wide
        atomicAdd(bar, 1u);
        while (__hip_atomic_load(bar, __ATOMIC_ACQUIRE, __HIP_MEMORY_SCOPE_AGENT)
               < target) {
            __builtin_amdgcn_s_sleep(2);
        }
    }
    __syncthreads();
}

// ---------------------------------------------------------------------------
// Persistent recurrent scan. Grid = 64 blocks x 128 threads (4 waves).
//   block  -> hidden n-tile (16 cols), waves -> batch m-tiles (4 x 16 rows)
// Each block parks its slice of W_hh (3 gates x 16 rows x 1024, padded) in
// LDS once via GLOBAL_LOAD_ASYNC_TO_LDS (ASYNCcnt-tracked, no VGPR round
// trip), then iterates all 512 timesteps reading B-operands from LDS
// (ds_load_b128, bank-conflict-free) and h from L2. Steps are separated by a
// device-wide software barrier.
//
//   r = sigmoid(xr + h@Whr^T + bhr)
//   z = sigmoid(xz + h@Whz^T + bhz)
//   n = tanh  (xn + r*(h@Whn^T + bhn))
//   h' = (1-z)*n + z*h
// ---------------------------------------------------------------------------
__global__ void __launch_bounds__(128) k_scan(
        const uint16_t* __restrict__ whh,      // [3H, H] bf16
        const float* __restrict__ bias_hh,     // [3H]
        const float* __restrict__ gates,       // [S][B][3H] fp32 time-major
        uint16_t* hb0, uint16_t* hb1,          // h bf16 ping/pong [B,H]
        float* hf0, float* hf1,                // h fp32 ping/pong [B,H]
        float* __restrict__ out,               // [B,S,H]
        unsigned int* bar) {
    extern __shared__ uint16_t s_whh[];        // 48 rows x 1032 halves

    const int lane  = threadIdx.x & 31;
    const int mtile = threadIdx.x >> 5;        // 0..3
    const int ntile = blockIdx.x;              // 0..63
    const int m0 = mtile * 16;
    const int n0 = ntile * 16;

    // --- stage this block's W_hh slice into LDS with async global->LDS DMA.
    // Dynamic LDS is the only allocation, so the byte index into s_whh is the
    // wave-relative LDS offset the instruction consumes. 48 async B128 ops
    // per wave (< ASYNCcnt capacity), drained with s_wait_asynccnt.
    for (int chunk = threadIdx.x; chunk < LDS_ROWS * (H_ / 8); chunk += 128) {
        int row = chunk >> 7;                  // 0..47  (H_/8 == 128 chunks/row)
        int c   = chunk & 127;
        int g   = row >> 4;
        int j   = row & 15;
        uint64_t gaddr  = (uint64_t)(whh + ((size_t)(g * H_ + n0 + j)) * H_ + c * 8);
        uint32_t ldsoff = (uint32_t)(row * LDS_ROW + c * 8) * 2u;
        asm volatile("global_load_async_to_lds_b128 %0, %1, off"
                     :
                     : "v"(ldsoff), "v"(gaddr)
                     : "memory");
    }
    asm volatile("s_wait_asynccnt 0x0" ::: "memory");
    __syncthreads();

    // per-lane constants
    const int n   = n0 + (lane & 15);
    const int mh  = (lane >> 4) << 3;
    const float bhr = bias_hh[n];
    const float bhz = bias_hh[H_ + n];
    const float bhn = bias_hh[2 * H_ + n];

    // this lane's B-tile base inside each gate's LDS slab
    const int bcol  = lane & 15;
    const int koffB = (lane >> 4) << 4;
    const uint16_t* lr = s_whh + (0 * 16 + bcol) * LDS_ROW + koffB;
    const uint16_t* lz = s_whh + (1 * 16 + bcol) * LDS_ROW + koffB;
    const uint16_t* ln = s_whh + (2 * 16 + bcol) * LDS_ROW + koffB;

    for (int t = 0; t < S_; ++t) {
        const uint16_t* hbin = (t & 1) ? hb1 : hb0;
        const float*    hfin = (t & 1) ? hf1 : hf0;
        uint16_t* hbout = (t & 1) ? hb0 : hb1;
        float*    hfout = (t & 1) ? hf0 : hf1;

        v8f aR = {}, aZ = {}, aN = {};
        for (int k0 = 0; k0 < H_; k0 += 32) {
            v16bf a  = load_tileA(hbin, m0, H_, k0, lane);
            v16bf br = lds_tileB(lr + k0);
            v16bf bz = lds_tileB(lz + k0);
            v16bf bn = lds_tileB(ln + k0);
            aR = WMMA_BF16(a, br, aR);
            aZ = WMMA_BF16(a, bz, aZ);
            aN = WMMA_BF16(a, bn, aN);
        }

        const float* gt = gates + (size_t)t * B_ * G_;
#pragma unroll
        for (int r = 0; r < 8; ++r) {
            const int bb = m0 + mh + r;
            const float* g = gt + (size_t)bb * G_;
            float xr = g[n];
            float xz = g[H_ + n];
            float xn = g[2 * H_ + n];
            float hprev = hfin[(size_t)bb * H_ + n];

            float rr = sigmoidf_(xr + aR[r] + bhr);
            float zz = sigmoidf_(xz + aZ[r] + bhz);
            float nn = tanhf(xn + rr * (aN[r] + bhn));
            float hnew = (1.0f - zz) * nn + zz * hprev;

            hfout[(size_t)bb * H_ + n] = hnew;
            hbout[(size_t)bb * H_ + n] = f2bf(hnew);
            out[((size_t)bb * S_ + t) * H_ + n] = hnew;
        }

        if (t + 1 < S_) grid_sync(bar, (unsigned int)(64 * (t + 1)));
    }
}

// ---------------------------------------------------------------------------
// Host-side launch
// ---------------------------------------------------------------------------
extern "C" void kernel_launch(void* const* d_in, const int* in_sizes, int n_in,
                              void* d_out, int out_size, void* d_ws, size_t ws_size,
                              hipStream_t stream) {
    const float* inputs = (const float*)d_in[0];   // [B,S,I]
    const float* h0     = (const float*)d_in[1];   // [1,B,H]
    const float* w_ih   = (const float*)d_in[2];   // [3H,I]
    const float* w_hh   = (const float*)d_in[3];   // [3H,H]
    const float* b_ih   = (const float*)d_in[4];   // [3H]
    const float* b_hh   = (const float*)d_in[5];   // [3H]
    float* out = (float*)d_out;                    // [B,S,H] ++ [1,B,H]

    // Workspace layout (bytes, 256-aligned by construction)
    char* ws = (char*)d_ws;
    size_t off = 0;
    float*    gates = (float*)(ws + off);    off += (size_t)S_ * B_ * G_ * 4;  // 402.7 MB
    uint16_t* xbf   = (uint16_t*)(ws + off); off += (size_t)B_ * S_ * I_ * 2;  //  33.6 MB
    uint16_t* wihb  = (uint16_t*)(ws + off); off += (size_t)G_ * I_ * 2;       //   3.1 MB
    uint16_t* whhb  = (uint16_t*)(ws + off); off += (size_t)G_ * H_ * 2;       //   6.3 MB
    float*    hf[2];
    hf[0] = (float*)(ws + off);              off += (size_t)B_ * H_ * 4;
    hf[1] = (float*)(ws + off);              off += (size_t)B_ * H_ * 4;
    uint16_t* hb[2];
    hb[0] = (uint16_t*)(ws + off);           off += (size_t)B_ * H_ * 2;
    hb[1] = (uint16_t*)(ws + off);           off += (size_t)B_ * H_ * 2;
    unsigned int* bar = (unsigned int*)(ws + off); off += 256;

    // 0) reset device barrier (required every launch: graph replay determinism)
    k_zero_u32<<<1, 64, 0, stream>>>(bar, 1);

    // 1) fp32 -> bf16 conversions + h0 init
    {
        int n = B_ * S_ * I_;
        k_f32_to_bf16<<<(n + 255) / 256, 256, 0, stream>>>(inputs, xbf, n);
        n = G_ * I_;
        k_f32_to_bf16<<<(n + 255) / 256, 256, 0, stream>>>(w_ih, wihb, n);
        n = G_ * H_;
        k_f32_to_bf16<<<(n + 255) / 256, 256, 0, stream>>>(w_hh, whhb, n);
        n = B_ * H_;
        k_init_h<<<(n + 255) / 256, 256, 0, stream>>>(h0, hf[0], hb[0], n);
    }

    // 2) input projection GEMM (WMMA bf16, fp32 accumulate)
    {
        const int mtiles = (B_ * S_) / 16;      // 2048
        const int ngrps  = G_ / 64;             // 48
        const int waves  = mtiles * ngrps;      // 98304
        k_xgemm<<<waves / 8, 256, 0, stream>>>(xbf, wihb, b_ih, gates);
    }

    // 3) persistent recurrent scan: 64 blocks, W_hh slice LDS-resident via
    //    async global->LDS, device-wide software barrier between timesteps
    k_scan<<<64, 128, LDS_BYTES, stream>>>(whhb, b_hh, gates,
                                           hb[0], hb[1], hf[0], hf[1],
                                           out, bar);

    // 4) final hidden state hn -> tail of d_out (after 512 steps: buffer 0)
    {
        int n = B_ * H_;
        k_copy_f32<<<(n + 255) / 256, 256, 0, stream>>>(hf[0], out + (size_t)B_ * S_ * H_, n);
    }
}
